// ILN_Interpolate_33655363731578
// MI455X (gfx1250) — matrix operand order repaired and checked by reading, not compiled, not run on hardware
//
#include <hip/hip_runtime.h>
#include <hip/hip_bf16.h>

// ---- problem constants (from reference) ----
#define Bn   4
#define Qn   32768
#define Cn   64
#define Hn   256
#define Wn   256
#define HWn  65536
#define NROW 131072   // B*Q

#define WAVES 4       // waves per block (128 threads)
#define SH 72         // f16 row stride (halfs) for activation panels
#define SX 68         // f32 row stride (floats) for residual panel

typedef __attribute__((ext_vector_type(16))) _Float16 v16h;
typedef __attribute__((ext_vector_type(8)))  _Float16 h8;
typedef __attribute__((ext_vector_type(8)))  float    v8f;

// ---------- WMMA fragment helpers ----------
// A fragment (16x32 f16) from an LDS panel, rows 0..15, row stride SH halfs.
// ISA layout: lane<16: v0..3 = K0..7, v4..7 = K16..23 (row = lane);
//             lane>=16: v0..3 = K8..15, v4..7 = K24..31 (row = lane-16).
__device__ __forceinline__ v16h load_Afrag(const _Float16* row0, int lane, int kh) {
  const int r   = lane & 15;
  const int hiK = (lane >> 4) & 1;
  const _Float16* p = row0 + r * SH + kh * 32 + hiK * 8;
  h8 lo = *(const h8*)(p);
  h8 hi = *(const h8*)(p + 16);
  v16h a;
#pragma unroll
  for (int i = 0; i < 8; ++i) { a[i] = lo[i]; a[i + 8] = hi[i]; }
  return a;
}

// B fragment (32x16 f16) from pre-transposed weights Wt[n][k] (n-major, 64 halfs/row).
// ISA layout: column N = lane%16; lanes 0..15 hold K=0..7,16..23, lanes 16..31 K=8..15,24..31
// (same K order as the A fragment).
__device__ __forceinline__ v16h load_Bfrag(const _Float16* __restrict__ Wt, int lane,
                                           int nt, int kh) {
  const int n  = nt * 16 + (lane & 15);
  const int k0 = kh * 32 + ((lane >> 4) & 1) * 8;
  const h8* p = (const h8*)(Wt + n * 64 + k0);
  h8 lo = p[0];
  h8 hi = p[2];
  v16h b;
#pragma unroll
  for (int i = 0; i < 8; ++i) { b[i] = lo[i]; b[i + 8] = hi[i]; }
  return b;
}

// One 16x16 output tile of (16x64 f16) @ (64x64 f16) + Cin, K reduced via 2 WMMAs.
__device__ __forceinline__ v8f mmtile(v16h a0, v16h a1, const _Float16* __restrict__ Wt,
                                      int lane, int nt, v8f cin) {
  v16h b0 = load_Bfrag(Wt, lane, nt, 0);
  v16h b1 = load_Bfrag(Wt, lane, nt, 1);
  v8f c = __builtin_amdgcn_wmma_f32_16x16x32_f16(false, a0, false, b0, (short)0, cin,
                                                 false, false);
  c = __builtin_amdgcn_wmma_f32_16x16x32_f16(false, a1, false, b1, (short)0, c,
                                             false, false);
  return c;
}

// ---------- prep: transpose+convert six 64x64 f32 weights to f16 Wt[n][k] ----------
__global__ void iln_prep_weights(const float* __restrict__ wq, const float* __restrict__ wk,
                                 const float* __restrict__ wv, const float* __restrict__ wo,
                                 const float* __restrict__ wm1, const float* __restrict__ wm2,
                                 _Float16* __restrict__ wt) {
  int i = blockIdx.x * blockDim.x + threadIdx.x;
  if (i >= 6 * 4096) return;
  int m = i >> 12, idx = i & 4095;
  int k = idx >> 6, n = idx & 63;
  const float* s = (m == 0) ? wq : (m == 1) ? wk : (m == 2) ? wv
                 : (m == 3) ? wo : (m == 4) ? wm1 : wm2;
  wt[m * 4096 + n * 64 + k] = (_Float16)s[k * 64 + n];
}

// ---------- fused main kernel: one wave = 4 queries = 16 token rows ----------
__global__ __launch_bounds__(128)
void iln_fused(const float* __restrict__ depth, const float* __restrict__ feat,
               const float* __restrict__ coord,
               const float* __restrict__ w_ce, const float* __restrict__ b_ce,
               const float* __restrict__ ln1_g, const float* __restrict__ ln1_b,
               const float* __restrict__ b_o,
               const float* __restrict__ ln2_g, const float* __restrict__ ln2_b,
               const float* __restrict__ b_m1, const float* __restrict__ b_m2,
               const float* __restrict__ w_head, const float* __restrict__ b_head,
               const _Float16* __restrict__ wt, float* __restrict__ out) {
  __shared__ __align__(16) float     XS[WAVES][16][SX];      // residual x (f32)
  __shared__ __align__(16) _Float16  HS[WAVES][16][SH];      // f16 A-source panel
  __shared__ __align__(16) _Float16  QKVS[WAVES][3][16][SH]; // q,k,v (f16); [0] reused as MLP mid
  __shared__ float PREDS[WAVES][16];
  __shared__ float LGTS[WAVES][16];

  const int wave  = threadIdx.x >> 5;
  const int lane  = threadIdx.x & 31;
  const int qbase = blockIdx.x * (WAVES * 4) + wave * 4;   // first query row of this wave

  _Float16* hrow0 = &HS[wave][0][0];
  const int ll  = lane & 15;
  const int hi8 = (lane >> 4) * 8;

  // ---- Step A: nearest-neighbor gather + rel-coord embedding ----
  const int r = lane >> 1, half = lane & 1;   // 2 lanes per token row
  const int nrow = qbase + (r >> 2);          // flattened b*Q+q
  const int t = r & 3;                        // (vx,vy): t bit1->vx, bit0->vy
  const int b = nrow >> 15;                   // / Qn
  const float cy0 = coord[nrow * 2 + 0];
  const float cx0 = coord[nrow * 2 + 1];
  const float vx = (t & 2) ? 1.f : -1.f;
  const float vy = (t & 1) ? 1.f : -1.f;
  const float rxy = 1.f / 256.f;
  float cy = fminf(fmaxf(cy0 + vx * rxy + 1e-6f, -1.f + 1e-6f), 1.f - 1e-6f);
  float cx = fminf(fmaxf(cx0 + vy * rxy + 1e-6f, -1.f + 1e-6f), 1.f - 1e-6f);
  int iy = (int)rintf((cy + 1.f) * 128.f - 0.5f); iy = min(max(iy, 0), 255);
  int ix = (int)rintf((cx + 1.f) * 128.f - 0.5f); ix = min(max(ix, 0), 255);
  const int lin = iy * Wn + ix;
  const float ycen = (float)(2 * iy + 1) * rxy - 1.f;
  const float xcen = (float)(2 * ix + 1) * rxy - 1.f;
  const float rel0 = (cy0 - ycen) * 256.f;
  const float rel1 = (cx0 - xcen) * 256.f;
  if (half == 0) PREDS[wave][r] = depth[(size_t)b * HWn + lin];

  float xr[32];
  const float* fb = feat + (size_t)b * Cn * HWn + lin;
  const int c0 = half * 32;
#pragma unroll 8
  for (int i = 0; i < 32; ++i) {
    const int c = c0 + i;
    float f = fb[(size_t)c * HWn];
    xr[i] = f + rel0 * w_ce[c] + rel1 * w_ce[64 + c] + b_ce[c];
    XS[wave][r][c] = xr[i];
  }

  // ---- Step B: LayerNorm 1 -> HS (f16) ----
  {
    float s = 0.f, s2 = 0.f;
#pragma unroll
    for (int i = 0; i < 32; ++i) { s += xr[i]; s2 += xr[i] * xr[i]; }
    s  += __shfl_xor(s, 1, 32);
    s2 += __shfl_xor(s2, 1, 32);
    const float mean = s * (1.f / 64.f);
    const float var  = s2 * (1.f / 64.f) - mean * mean;
    const float inv  = rsqrtf(var + 1e-5f);
#pragma unroll 8
    for (int i = 0; i < 32; ++i) {
      const int c = c0 + i;
      HS[wave][r][c] = (_Float16)((xr[i] - mean) * inv * ln1_g[c] + ln1_b[c]);
    }
  }
  __syncthreads();

  // ---- Step C: Q,K,V = LN1 @ Wq/Wk/Wv (WMMA) ----
  {
    v16h a0 = load_Afrag(hrow0, lane, 0);
    v16h a1 = load_Afrag(hrow0, lane, 1);
#pragma unroll
    for (int w = 0; w < 3; ++w) {
      const _Float16* Wm = wt + w * 4096;
#pragma unroll
      for (int nt = 0; nt < 4; ++nt) {
        v8f cin = {};
        v8f c = mmtile(a0, a1, Wm, lane, nt, cin);
#pragma unroll
        for (int j = 0; j < 8; ++j)
          QKVS[wave][w][j + hi8][nt * 16 + ll] = (_Float16)c[j];
      }
    }
  }
  __syncthreads();

  // ---- Step D: 8-head attention over seq=4 (lane = (qlocal, head)) -> o into HS ----
  {
    const int ql = lane >> 3, hd = lane & 7;
    const int rb = ql * 4, cb = hd * 8;
    float qm[4][8], km[4][8], vm[4][8];
#pragma unroll
    for (int tt = 0; tt < 4; ++tt)
#pragma unroll
      for (int d = 0; d < 8; ++d) {
        qm[tt][d] = (float)QKVS[wave][0][rb + tt][cb + d];
        km[tt][d] = (float)QKVS[wave][1][rb + tt][cb + d];
        vm[tt][d] = (float)QKVS[wave][2][rb + tt][cb + d];
      }
    const float scale = 0.35355339059327373f;  // 1/sqrt(8)
#pragma unroll
    for (int tt = 0; tt < 4; ++tt) {
      float sc[4];
#pragma unroll
      for (int ss = 0; ss < 4; ++ss) {
        float d0 = 0.f;
#pragma unroll
        for (int d = 0; d < 8; ++d) d0 += qm[tt][d] * km[ss][d];
        sc[ss] = d0 * scale;
      }
      float mx = fmaxf(fmaxf(sc[0], sc[1]), fmaxf(sc[2], sc[3]));
      float e[4], den = 0.f;
#pragma unroll
      for (int ss = 0; ss < 4; ++ss) { e[ss] = __expf(sc[ss] - mx); den += e[ss]; }
      const float rden = 1.f / den;
#pragma unroll
      for (int d = 0; d < 8; ++d) {
        float o = 0.f;
#pragma unroll
        for (int ss = 0; ss < 4; ++ss) o += e[ss] * rden * vm[ss][d];
        HS[wave][rb + tt][cb + d] = (_Float16)o;
      }
    }
  }
  __syncthreads();

  // ---- Step E: x = x + o @ Wo + b_o (WMMA, C-init from residual) ----
  {
    v16h a0 = load_Afrag(hrow0, lane, 0);
    v16h a1 = load_Afrag(hrow0, lane, 1);
    const _Float16* Wm = wt + 3 * 4096;
#pragma unroll
    for (int nt = 0; nt < 4; ++nt) {
      const int col = nt * 16 + ll;
      const float bb = b_o[col];
      v8f cin;
#pragma unroll
      for (int j = 0; j < 8; ++j) cin[j] = XS[wave][j + hi8][col] + bb;
      v8f c = mmtile(a0, a1, Wm, lane, nt, cin);
#pragma unroll
      for (int j = 0; j < 8; ++j) XS[wave][j + hi8][col] = c[j];
    }
  }
  __syncthreads();

  // ---- Step F: LayerNorm 2 -> HS (f16) ----
  {
    float y[32], s = 0.f, s2 = 0.f;
#pragma unroll 8
    for (int i = 0; i < 32; ++i) {
      y[i] = XS[wave][r][c0 + i];
      s += y[i]; s2 += y[i] * y[i];
    }
    s  += __shfl_xor(s, 1, 32);
    s2 += __shfl_xor(s2, 1, 32);
    const float mean = s * (1.f / 64.f);
    const float var  = s2 * (1.f / 64.f) - mean * mean;
    const float inv  = rsqrtf(var + 1e-5f);
#pragma unroll 8
    for (int i = 0; i < 32; ++i) {
      const int c = c0 + i;
      HS[wave][r][c] = (_Float16)((y[i] - mean) * inv * ln2_g[c] + ln2_b[c]);
    }
  }
  __syncthreads();

  // ---- Step G: m = gelu(LN2 @ Wm1 + b_m1) -> QKVS[wave][0] (f16) ----
  {
    v16h a0 = load_Afrag(hrow0, lane, 0);
    v16h a1 = load_Afrag(hrow0, lane, 1);
    const _Float16* Wm = wt + 4 * 4096;
#pragma unroll
    for (int nt = 0; nt < 4; ++nt) {
      const int col = nt * 16 + ll;
      const float bb = b_m1[col];
      v8f cin;
#pragma unroll
      for (int j = 0; j < 8; ++j) cin[j] = bb;
      v8f c = mmtile(a0, a1, Wm, lane, nt, cin);
#pragma unroll
      for (int j = 0; j < 8; ++j) {
        float u = c[j];
        float g = 0.5f * u * (1.f + tanhf(0.7978845608f * (u + 0.044715f * u * u * u)));
        QKVS[wave][0][j + hi8][col] = (_Float16)g;
      }
    }
  }
  __syncthreads();

  // ---- Step H: x = x + m @ Wm2 + b_m2 (WMMA) ----
  {
    const _Float16* mrow0 = &QKVS[wave][0][0][0];
    v16h a0 = load_Afrag(mrow0, lane, 0);
    v16h a1 = load_Afrag(mrow0, lane, 1);
    const _Float16* Wm = wt + 5 * 4096;
#pragma unroll
    for (int nt = 0; nt < 4; ++nt) {
      const int col = nt * 16 + ll;
      const float bb = b_m2[col];
      v8f cin;
#pragma unroll
      for (int j = 0; j < 8; ++j) cin[j] = XS[wave][j + hi8][col] + bb;
      v8f c = mmtile(a0, a1, Wm, lane, nt, cin);
#pragma unroll
      for (int j = 0; j < 8; ++j) XS[wave][j + hi8][col] = c[j];
    }
  }
  __syncthreads();

  // ---- Step I: head logits, softmax over 4 tokens, weighted depth sum ----
  {
    float dot = 0.f;
#pragma unroll 8
    for (int i = 0; i < 32; ++i) dot += XS[wave][r][c0 + i] * w_head[c0 + i];
    dot += __shfl_xor(dot, 1, 32);
    if (half == 0) LGTS[wave][r] = dot + b_head[0];
  }
  __syncthreads();

  if (lane < 4) {
    float l[4], p[4];
#pragma unroll
    for (int tt = 0; tt < 4; ++tt) {
      l[tt] = LGTS[wave][lane * 4 + tt];
      p[tt] = PREDS[wave][lane * 4 + tt];
    }
    float mx = fmaxf(fmaxf(l[0], l[1]), fmaxf(l[2], l[3]));
    float den = 0.f, acc = 0.f;
#pragma unroll
    for (int tt = 0; tt < 4; ++tt) {
      float e = __expf(l[tt] - mx);
      den += e;
      acc += e * p[tt];
    }
    out[qbase + lane] = acc / den;
  }
}

extern "C" void kernel_launch(void* const* d_in, const int* in_sizes, int n_in,
                              void* d_out, int out_size, void* d_ws, size_t ws_size,
                              hipStream_t stream) {
  const float* depth  = (const float*)d_in[0];
  const float* feat   = (const float*)d_in[1];
  const float* coord  = (const float*)d_in[2];
  const float* w_ce   = (const float*)d_in[3];
  const float* b_ce   = (const float*)d_in[4];
  const float* ln1_g  = (const float*)d_in[5];
  const float* ln1_b  = (const float*)d_in[6];
  const float* w_q    = (const float*)d_in[7];
  const float* w_k    = (const float*)d_in[8];
  const float* w_v    = (const float*)d_in[9];
  const float* w_o    = (const float*)d_in[10];
  const float* b_o    = (const float*)d_in[11];
  const float* ln2_g  = (const float*)d_in[12];
  const float* ln2_b  = (const float*)d_in[13];
  const float* w_m1   = (const float*)d_in[14];
  const float* b_m1   = (const float*)d_in[15];
  const float* w_m2   = (const float*)d_in[16];
  const float* b_m2   = (const float*)d_in[17];
  const float* w_head = (const float*)d_in[18];
  const float* b_head = (const float*)d_in[19];
  float* out = (float*)d_out;
  _Float16* wt = (_Float16*)d_ws;   // 6 * 4096 halfs = 48 KB

  iln_prep_weights<<<(6 * 4096 + 127) / 128, 128, 0, stream>>>(
      w_q, w_k, w_v, w_o, w_m1, w_m2, wt);

  // 8192 blocks * (4 waves * 4 queries) = 131072 query rows, exact cover
  iln_fused<<<NROW / (WAVES * 4), 128, 0, stream>>>(
      depth, feat, coord, w_ce, b_ce, ln1_g, ln1_b, b_o, ln2_g, ln2_b,
      b_m1, b_m2, w_head, b_head, wt, out);
}
